// BertSelfAttention_37692632990185
// MI455X (gfx1250) — matrix operand order repaired and checked by reading, not compiled, not run on hardware
//
#include <hip/hip_runtime.h>
#include <hip/hip_bf16.h>

#define B_   2
#define S_   2048
#define HID_ 1024
#define NH_  16
#define HD_  64

typedef __attribute__((ext_vector_type(16))) __bf16 v16bf;
typedef __attribute__((ext_vector_type(8)))  __bf16 v8bf;
typedef __attribute__((ext_vector_type(8)))  float  v8f;

__device__ __forceinline__ v8f wmma_bf16(v16bf a, v16bf b, v8f c) {
  // (neg_a, A, neg_b, B, c_mod, C, reuse_a, reuse_b)
  return __builtin_amdgcn_wmma_f32_16x16x32_bf16(false, a, false, b, (short)0, c, false, false);
}

// A fragment: 16x32 bf16, row-major source (ld in elements).
// Lane group 0 (lanes 0-15): row=lane, K = {0..7, 16..23}; group 1: K = {8..15, 24..31}.
__device__ __forceinline__ v16bf load_a_frag(const __bf16* base, int ld, int lane) {
  const int r  = lane & 15;
  const int ko = (lane >> 4) << 3;            // 0 or 8
  const __bf16* p = base + (size_t)r * ld + ko;
  v8bf lo = *(const v8bf*)(p);                // K = ko .. ko+7
  v8bf hi = *(const v8bf*)(p + 16);           // K = 16+ko .. 16+ko+7
  return __builtin_shufflevector(lo, hi, 0,1,2,3,4,5,6,7,8,9,10,11,12,13,14,15);
}

// B fragment: 32x16 bf16 (K x N) where storage row c holds column c of B
// (i.e. K-contiguous). Lane group 0: K = {0..7, 16..23}; group 1: K = {8..15, 24..31}.
__device__ __forceinline__ v16bf load_b_frag(const __bf16* base, int ld, int lane) {
  const int c  = lane & 15;
  const int ks = (lane >> 4) << 3;            // 0 or 8
  const __bf16* p = base + (size_t)c * ld + ks;
  v8bf lo = *(const v8bf*)(p);
  v8bf hi = *(const v8bf*)(p + 16);
  return __builtin_shufflevector(lo, hi, 0,1,2,3,4,5,6,7,8,9,10,11,12,13,14,15);
}

__global__ void __launch_bounds__(256) cvt_f32_bf16(const float* __restrict__ src,
                                                    __bf16* __restrict__ dst, int n) {
  int i = blockIdx.x * blockDim.x + threadIdx.x;
  int stride = gridDim.x * blockDim.x;
  for (; i < n; i += stride) dst[i] = (__bf16)src[i];
}

// One wave computes a 32(M) x 32(N) tile of out = X @ W^T + b in bf16,
// writing Q (pre-scaled by 1/sqrt(HD)) and K as [B,NH,S,HD], V transposed as [B,NH,HD,S].
__global__ void __launch_bounds__(128) qkv_gemm(
    const __bf16* __restrict__ X,
    const __bf16* __restrict__ Wq, const __bf16* __restrict__ Wk, const __bf16* __restrict__ Wv,
    const float* __restrict__ bq, const float* __restrict__ bk, const float* __restrict__ bv,
    __bf16* __restrict__ Qo, __bf16* __restrict__ Ko, __bf16* __restrict__ Vto) {
  const int lane = threadIdx.x & 31;
  const int wid  = (blockIdx.x << 2) + (threadIdx.x >> 5);   // 0..4095
  const int mat  = blockIdx.z;                               // 0=Q 1=K 2=V
  const int nt = wid & 31, mt = wid >> 5;
  const int m0 = mt * 32, n0 = nt * 32;

  const __bf16* W    = (mat == 0) ? Wq : (mat == 1) ? Wk : Wv;
  const float*  bias = (mat == 0) ? bq : (mat == 1) ? bk : bv;
  __bf16*       out  = (mat == 0) ? Qo : (mat == 1) ? Ko : Vto;
  const bool    vt   = (mat == 2);
  const float   oscale = (mat == 0) ? 0.125f : 1.0f;         // fold 1/sqrt(64) into Q

  v8f acc[2][2] = {};
  for (int k = 0; k < HID_; k += 32) {
    v16bf a0 = load_a_frag(X + (size_t)m0        * HID_ + k, HID_, lane);
    v16bf a1 = load_a_frag(X + (size_t)(m0 + 16) * HID_ + k, HID_, lane);
    v16bf b0 = load_b_frag(W + (size_t)n0        * HID_ + k, HID_, lane);
    v16bf b1 = load_b_frag(W + (size_t)(n0 + 16) * HID_ + k, HID_, lane);
    acc[0][0] = wmma_bf16(a0, b0, acc[0][0]);
    acc[0][1] = wmma_bf16(a0, b1, acc[0][1]);
    acc[1][0] = wmma_bf16(a1, b0, acc[1][0]);
    acc[1][1] = wmma_bf16(a1, b1, acc[1][1]);
  }

  const int rbase = (lane < 16) ? 0 : 8;
  const int cl = lane & 15;
#pragma unroll
  for (int mi = 0; mi < 2; ++mi)
#pragma unroll
    for (int ni = 0; ni < 2; ++ni) {
      int col = n0 + ni * 16 + cl;
      int h = col >> 6, d = col & 63;
      float bval = bias[col];
#pragma unroll
      for (int r = 0; r < 8; ++r) {
        int row = m0 + mi * 16 + rbase + r;
        int bb = row >> 11, ss = row & (S_ - 1);
        float val = (acc[mi][ni][r] + bval) * oscale;
        size_t idx = vt ? (((size_t)(bb * NH_ + h) * HD_ + d) * S_ + ss)
                        : (((size_t)(bb * NH_ + h) * S_ + ss) * HD_ + d);
        out[idx] = (__bf16)val;
      }
    }
}

// Flash attention: one wave per 16-row query tile, streaming 32 keys/iteration.
// Q is pre-scaled by 1/sqrt(HD); softmax rows are split across lane pairs.
__global__ void __launch_bounds__(128) flash_attn(
    const __bf16* __restrict__ Q, const __bf16* __restrict__ K,
    const __bf16* __restrict__ Vt, const float* __restrict__ mask,
    float* __restrict__ out) {
  __shared__ __attribute__((aligned(16))) float  sS[4][16][32];
  __shared__ __attribute__((aligned(16))) __bf16 sP[4][16][32];
  __shared__ float sAlpha[4][16];
  __shared__ float sLinv[4][16];

  const int lane = threadIdx.x & 31;
  const int w  = threadIdx.x >> 5;
  const int id = (blockIdx.x << 2) + w;        // 0..4095 = B*NH*(S/16)
  const int qt = id & 127;
  const int h  = (id >> 7) & 15;
  const int b  = id >> 11;
  const int q0 = qt << 4;

  const __bf16* Qb = Q  + ((size_t)(b * NH_ + h) * S_ + q0) * HD_;
  const __bf16* Kb = K  + ((size_t)(b * NH_ + h) * S_) * HD_;
  const __bf16* Vb = Vt + ((size_t)(b * NH_ + h) * HD_) * S_;   // [HD][S]
  const float*  Mk = mask + ((size_t)b * S_ + q0) * S_;

  v16bf qa0 = load_a_frag(Qb + 0,  HD_, lane);   // d = 0..31
  v16bf qa1 = load_a_frag(Qb + 32, HD_, lane);   // d = 32..63

  v8f ctx[4] = {};
  float m_run = -1e30f, l_run = 0.0f;            // stats for row (lane>>1)
  const int rbase = (lane < 16) ? 0 : 8;         // C-fragment row base
  const int cl    = lane & 15;                   // C-fragment column
  const int srow  = lane >> 1;                   // softmax row owned by lane pair
  const int shalf = (lane & 1) << 4;             // 0 or 16: column half of that row

  for (int kb = 0; kb < S_; kb += 32) {
    // ---- scores: S[16 x 32] = Qs @ K^T, two 16x16 C tiles ----
    v8f s0 = {}, s1 = {};
    {
      v16bf b00 = load_b_frag(Kb + (size_t)kb * HD_ + 0,  HD_, lane);
      v16bf b01 = load_b_frag(Kb + (size_t)kb * HD_ + 32, HD_, lane);
      s0 = wmma_bf16(qa0, b00, s0);
      s0 = wmma_bf16(qa1, b01, s0);
      v16bf b10 = load_b_frag(Kb + (size_t)(kb + 16) * HD_ + 0,  HD_, lane);
      v16bf b11 = load_b_frag(Kb + (size_t)(kb + 16) * HD_ + 32, HD_, lane);
      s1 = wmma_bf16(qa0, b10, s1);
      s1 = wmma_bf16(qa1, b11, s1);
    }
    // park raw scores in per-wave LDS (C-fragment layout -> row-major tile)
#pragma unroll
    for (int r = 0; r < 8; ++r) {
      sS[w][rbase + r][cl]      = s0[r];
      sS[w][rbase + r][16 + cl] = s1[r];
    }
    __syncthreads();
    // ---- online softmax: lane pair owns one row, 16 columns each ----
    const float* Mrow = Mk + (size_t)srow * S_ + kb + shalf;  // 16 contiguous floats
    float tmp[16];
    float rmax = -1e30f;
#pragma unroll
    for (int j = 0; j < 16; ++j) {
      tmp[j] = sS[w][srow][shalf + j] + (1.0f - Mrow[j]) * -10000.0f;
      rmax = fmaxf(rmax, tmp[j]);
    }
    rmax = fmaxf(rmax, __shfl_xor(rmax, 1));
    float newm  = fmaxf(m_run, rmax);
    float alpha = __expf(m_run - newm);
    float rsum  = 0.0f;
#pragma unroll
    for (int j = 0; j < 16; ++j) {
      float p = __expf(tmp[j] - newm);
      rsum += p;
      sP[w][srow][shalf + j] = (__bf16)p;
    }
    rsum += __shfl_xor(rsum, 1);
    m_run = newm;
    l_run = l_run * alpha + rsum;
    if ((lane & 1) == 0) sAlpha[w][srow] = alpha;
    __syncthreads();
    // ---- rescale accumulators, then ctx += P @ V (4 N-tiles of 16) ----
    float al[8];
#pragma unroll
    for (int r = 0; r < 8; ++r) al[r] = sAlpha[w][rbase + r];
#pragma unroll
    for (int t = 0; t < 4; ++t)
#pragma unroll
      for (int r = 0; r < 8; ++r) ctx[t][r] *= al[r];
    v16bf pa = load_a_frag(&sP[w][0][0], 32, lane);   // 16x32 probs
#pragma unroll
    for (int t = 0; t < 4; ++t) {
      v16bf vb = load_b_frag(Vb + (size_t)(t * 16) * S_ + kb, S_, lane);
      ctx[t] = wmma_bf16(pa, vb, ctx[t]);
    }
    __syncthreads();
  }

  if ((lane & 1) == 0) sLinv[w][srow] = 1.0f / l_run;
  __syncthreads();
  float li[8];
#pragma unroll
  for (int r = 0; r < 8; ++r) li[r] = sLinv[w][rbase + r];
  float* op = out + ((size_t)b * S_ + q0) * HID_ + h * HD_;
#pragma unroll
  for (int t = 0; t < 4; ++t)
#pragma unroll
    for (int r = 0; r < 8; ++r)
      op[(size_t)(rbase + r) * HID_ + t * 16 + cl] = ctx[t][r] * li[r];
}

extern "C" void kernel_launch(void* const* d_in, const int* in_sizes, int n_in,
                              void* d_out, int out_size, void* d_ws, size_t ws_size,
                              hipStream_t stream) {
  const float* hs   = (const float*)d_in[0];
  const float* mask = (const float*)d_in[1];
  const float* Wq   = (const float*)d_in[2];
  const float* bq   = (const float*)d_in[3];
  const float* Wk   = (const float*)d_in[4];
  const float* bk   = (const float*)d_in[5];
  const float* Wv   = (const float*)d_in[6];
  const float* bv   = (const float*)d_in[7];
  float* out = (float*)d_out;

  const size_t nX = (size_t)B_ * S_ * HID_;      // 4,194,304
  const size_t nW = (size_t)HID_ * HID_;         // 1,048,576
  const size_t nQ = (size_t)B_ * NH_ * S_ * HD_; // 4,194,304

  char* ws = (char*)d_ws;
  __bf16* Xb  = (__bf16*)ws; ws += nX * 2;
  __bf16* Wqb = (__bf16*)ws; ws += nW * 2;
  __bf16* Wkb = (__bf16*)ws; ws += nW * 2;
  __bf16* Wvb = (__bf16*)ws; ws += nW * 2;
  __bf16* Qb  = (__bf16*)ws; ws += nQ * 2;
  __bf16* Kb  = (__bf16*)ws; ws += nQ * 2;
  __bf16* Vtb = (__bf16*)ws; ws += nQ * 2;

  cvt_f32_bf16<<<2048, 256, 0, stream>>>(hs, Xb, (int)nX);
  cvt_f32_bf16<<<1024, 256, 0, stream>>>(Wq, Wqb, (int)nW);
  cvt_f32_bf16<<<1024, 256, 0, stream>>>(Wk, Wkb, (int)nW);
  cvt_f32_bf16<<<1024, 256, 0, stream>>>(Wv, Wvb, (int)nW);

  // 4096 wave-tiles per matrix (128 M-tiles x 32 N-tiles), 4 waves per block, z = {Q,K,V}
  qkv_gemm<<<dim3(1024, 1, 3), 128, 0, stream>>>(Xb, Wqb, Wkb, Wvb, bq, bk, bv, Qb, Kb, Vtb);

  // 4096 query tiles (B*NH*S/16), 4 waves per block
  flash_attn<<<1024, 128, 0, stream>>>(Qb, Kb, Vtb, mask, out);
}
